// SoftTripleLoss_39694087750018
// MI455X (gfx1250) — matrix operand order repaired and checked by reading, not compiled, not run on hardware
//
#include <hip/hip_runtime.h>
#include <hip/hip_bf16.h>

// ---------------------------------------------------------------------------
// SoftTriple loss, fused for MI455X (gfx1250, wave32, WMMA).
//   B=256, E=512, C=8192, K=10  (N = C*K = 81920 centers)
// Main GEMM (256 x 81920 x 512) runs on V_WMMA_F32_16X16X32_BF16 using the
// bf16x3 split (hi/lo) for near-f32 accuracy at ~8x the FLOP density of the
// f32 WMMA path. fc row norms are hoisted: x and the gram are scaled by
// invnorm instead of materializing normalized w (saves a 336 MB pass).
// ---------------------------------------------------------------------------

typedef __bf16 bf16_t;
typedef bf16_t v4bf  __attribute__((ext_vector_type(4)));
typedef bf16_t v8bf  __attribute__((ext_vector_type(8)));
typedef bf16_t v16bf __attribute__((ext_vector_type(16)));
typedef float  v8f   __attribute__((ext_vector_type(8)));

#define BB      256
#define EE      512
#define CC      8192
#define KK      10
#define NCTR    (CC * KK)          // 81920
#define LMD     10.0f
#define INV_GAMMA 10.0f
#define TAU     0.2f
#define MARGIN  0.01f

// ws layout (floats)
#define WS_INV   0                         // invnorm[81920]
#define WS_LOG   (NCTR)                    // logits[256*8192]
#define WS_ACC   (NCTR + BB * CC)          // acc[0]=loss, acc[1]=reg-sum

// ---------------------------------------------------------------------------
__global__ __launch_bounds__(32) void st_zero_kernel(float* acc) {
    if (threadIdx.x < 2) acc[threadIdx.x] = 0.0f;
}

// ---------------------------------------------------------------------------
// invnorm[row] = 1 / max(||fc_row||, 1e-12)   (one wave per row, strided)
// ---------------------------------------------------------------------------
__global__ __launch_bounds__(256) void st_invnorm_kernel(const float* __restrict__ fc,
                                                         float* __restrict__ invnorm) {
    const int ln = threadIdx.x & 31;
    const int gw = blockIdx.x * 8 + (threadIdx.x >> 5);   // 320 blocks * 8 waves
    for (int row = gw; row < NCTR; row += 2560) {
        const float4* rp = (const float4*)(fc + (size_t)row * EE);
        float s = 0.0f;
#pragma unroll
        for (int t = 0; t < 4; ++t) {
            float4 v = rp[ln + 32 * t];
            s += v.x * v.x + v.y * v.y + v.z * v.z + v.w * v.w;
        }
#pragma unroll
        for (int m = 16; m >= 1; m >>= 1) s += __shfl_xor(s, m, 32);
        if (ln == 0) invnorm[row] = 1.0f / fmaxf(sqrtf(s), 1e-12f);
    }
}

// ---------------------------------------------------------------------------
// Fused GEMM (bf16x3 on V_WMMA_F32_16X16X32_BF16) + softmax-pool + logits.
//   grid: (C/8, B/128); block: 320 threads = 10 waves.
//   WG tile: 80 centers x 128 batch = 5 M-subtiles x 8 N-subtiles.
//   wave w: mt = w%5, covers nt = (w/5)*4 .. +3  (4 accumulators).
// ---------------------------------------------------------------------------
#define KC  64                     // k-chunk staged in LDS (2 WMMA k-steps)
#define LDB 72                     // LDS row stride in halves (16B-aligned rows)
#define NBB 128                    // batch tile per WG

__global__ __launch_bounds__(320) void st_gemm_pool_kernel(const float* __restrict__ emb,
                                                           const int*   __restrict__ labels,
                                                           const float* __restrict__ fc,
                                                           const float* __restrict__ invnorm,
                                                           float* __restrict__ logits) {
    __shared__ bf16_t sFh[80 * LDB],  sFl[80 * LDB];    // fc chunk hi/lo [80 x KC]
    __shared__ bf16_t sEh[NBB * LDB], sEl[NBB * LDB];   // emb chunk hi/lo [128 x KC]
    __shared__ float  sX[80 * (NBB + 1)];               // x tile [center][batch]
    __shared__ float  sInv[80];

    const int tid   = threadIdx.x;
    const int lane  = tid & 31;
    const int half  = lane >> 4;
    const int lr    = lane & 15;
    const int wave  = tid >> 5;        // 0..9
    const int mt    = wave % 5;        // center sub-tile
    const int ntb   = (wave / 5) * 4;  // first of 4 batch sub-tiles

    const int cls0 = blockIdx.x * 8;
    const int r0   = cls0 * KK;
    const int b0   = blockIdx.y * NBB;

    if (tid < 80) sInv[tid] = invnorm[r0 + tid];

    v8f acc[4] = {};
    for (int ch = 0; ch < EE / KC; ++ch) {
        const int kc0 = ch * KC;
        __syncthreads();
        // stage fc chunk: split f32 -> bf16 hi + lo during the copy
        for (int idx = tid; idx < 80 * 16; idx += 320) {
            const int r = idx >> 4, c4 = (idx & 15) << 2;
            const float* gp = fc + (size_t)(r0 + r) * EE + kc0 + c4;
            if (ch + 1 < EE / KC) __builtin_prefetch(gp + KC, 0, 0);
            const float4 v = *(const float4*)gp;
            v4bf h, l;
            h.x = (bf16_t)v.x; l.x = (bf16_t)(v.x - (float)h.x);
            h.y = (bf16_t)v.y; l.y = (bf16_t)(v.y - (float)h.y);
            h.z = (bf16_t)v.z; l.z = (bf16_t)(v.z - (float)h.z);
            h.w = (bf16_t)v.w; l.w = (bf16_t)(v.w - (float)h.w);
            *(v4bf*)(&sFh[r * LDB + c4]) = h;
            *(v4bf*)(&sFl[r * LDB + c4]) = l;
        }
        // stage emb chunk, same split
        for (int idx = tid; idx < NBB * 16; idx += 320) {
            const int r = idx >> 4, c4 = (idx & 15) << 2;
            const float4 v = *(const float4*)(emb + (size_t)(b0 + r) * EE + kc0 + c4);
            v4bf h, l;
            h.x = (bf16_t)v.x; l.x = (bf16_t)(v.x - (float)h.x);
            h.y = (bf16_t)v.y; l.y = (bf16_t)(v.y - (float)h.y);
            h.z = (bf16_t)v.z; l.z = (bf16_t)(v.z - (float)h.z);
            h.w = (bf16_t)v.w; l.w = (bf16_t)(v.w - (float)h.w);
            *(v4bf*)(&sEh[r * LDB + c4]) = h;
            *(v4bf*)(&sEl[r * LDB + c4]) = l;
        }
        __syncthreads();
#pragma unroll
        for (int ks = 0; ks < KC / 32; ++ks) {
            const int k0 = ks * 32;
            // A fragment (16x32 bf16): halves [k0+8h,+8) and [k0+16+8h,+8)
            const int aoff = (mt * 16 + lr) * LDB + k0 + 8 * half;
            const v8bf ah0 = *(const v8bf*)(&sFh[aoff]);
            const v8bf ah1 = *(const v8bf*)(&sFh[aoff + 16]);
            const v8bf al0 = *(const v8bf*)(&sFl[aoff]);
            const v8bf al1 = *(const v8bf*)(&sFl[aoff + 16]);
            const v16bf ah = __builtin_shufflevector(ah0, ah1, 0,1,2,3,4,5,6,7,8,9,10,11,12,13,14,15);
            const v16bf al = __builtin_shufflevector(al0, al1, 0,1,2,3,4,5,6,7,8,9,10,11,12,13,14,15);
#pragma unroll
            for (int j = 0; j < 4; ++j) {
                // B fragment (32x16 bf16): halves [k0+8h,+8) and [k0+16+8h,+8)
                const int boff = ((ntb + j) * 16 + lr) * LDB + k0 + 8 * half;
                const v8bf bh0 = *(const v8bf*)(&sEh[boff]);
                const v8bf bh1 = *(const v8bf*)(&sEh[boff + 16]);
                const v8bf bl0 = *(const v8bf*)(&sEl[boff]);
                const v8bf bl1 = *(const v8bf*)(&sEl[boff + 16]);
                const v16bf bh = __builtin_shufflevector(bh0, bh1, 0,1,2,3,4,5,6,7,8,9,10,11,12,13,14,15);
                const v16bf bl = __builtin_shufflevector(bl0, bl1, 0,1,2,3,4,5,6,7,8,9,10,11,12,13,14,15);
                // bf16x3: ab ~= ah*bh + ah*bl + al*bh
                acc[j] = __builtin_amdgcn_wmma_f32_16x16x32_bf16(false, ah, false, bh,
                                                                 (short)0, acc[j], false, false);
                acc[j] = __builtin_amdgcn_wmma_f32_16x16x32_bf16(false, ah, false, bl,
                                                                 (short)0, acc[j], false, false);
                acc[j] = __builtin_amdgcn_wmma_f32_16x16x32_bf16(false, al, false, bh,
                                                                 (short)0, acc[j], false, false);
            }
        }
    }

    // scale by invnorm, park tiles in sX [center][batch]
#pragma unroll
    for (int j = 0; j < 4; ++j) {
        const int colC = (ntb + j) * 16 + lr;
#pragma unroll
        for (int v = 0; v < 8; ++v) {
            const int rowC = mt * 16 + v + 8 * half;
            sX[rowC * (NBB + 1) + colC] = acc[j][v] * sInv[rowC];
        }
    }
    __syncthreads();

    // softmax-pool over K=10 per (class, batch): 8*128 = 1024 pairs
    for (int idx = tid; idx < 8 * NBB; idx += 320) {
        const int b  = idx & (NBB - 1);
        const int cl = idx >> 7;
        float xv[KK];
        float mx = -1e30f;
#pragma unroll
        for (int k = 0; k < KK; ++k) {
            xv[k] = sX[(cl * KK + k) * (NBB + 1) + b];
            mx = fmaxf(mx, xv[k]);
        }
        float s = 0.0f, hn = 0.0f;
#pragma unroll
        for (int k = 0; k < KK; ++k) {
            const float e = __expf(INV_GAMMA * (xv[k] - mx));
            s += e;
            hn += e * xv[k];
        }
        const float h  = hn / s;
        const int  gc  = cls0 + cl;
        const int  gb  = b0 + b;
        const float mg = (labels[gb] == gc) ? MARGIN : 0.0f;
        logits[(size_t)gb * CC + gc] = LMD * (h - mg);
    }
}

// ---------------------------------------------------------------------------
// per-row cross entropy: loss_b = logsumexp(row) - row[label]; mean via atomics
// ---------------------------------------------------------------------------
__global__ __launch_bounds__(256) void st_ce_kernel(const float* __restrict__ logits,
                                                    const int*   __restrict__ labels,
                                                    float* __restrict__ acc) {
    __shared__ float red[256];
    const int row = blockIdx.x;
    const int tid = threadIdx.x;
    const float* lg = logits + (size_t)row * CC;

    float mx = -1e30f;
    for (int c = tid; c < CC; c += 256) mx = fmaxf(mx, lg[c]);
    red[tid] = mx;
    __syncthreads();
    for (int s = 128; s >= 1; s >>= 1) {
        if (tid < s) red[tid] = fmaxf(red[tid], red[tid + s]);
        __syncthreads();
    }
    mx = red[0];
    __syncthreads();

    float sum = 0.0f;
    for (int c = tid; c < CC; c += 256) sum += __expf(lg[c] - mx);
    red[tid] = sum;
    __syncthreads();
    for (int s = 128; s >= 1; s >>= 1) {
        if (tid < s) red[tid] += red[tid + s];
        __syncthreads();
    }
    if (tid == 0) {
        const float loss_b = logf(red[0]) + mx - lg[labels[row]];
        atomicAdd(&acc[0], loss_b * (1.0f / (float)BB));
    }
}

// ---------------------------------------------------------------------------
// regularizer: per class, 45 upper-tri pairs of the 10x10 normalized gram.
// ---------------------------------------------------------------------------
__global__ __launch_bounds__(256) void st_reg_kernel(const float* __restrict__ fc,
                                                     const float* __restrict__ invnorm,
                                                     float* __restrict__ acc) {
    __shared__ float rows[KK * EE];
    __shared__ float sinv[KK];
    const int c   = blockIdx.x;
    const int tid = threadIdx.x;
    const int wv  = tid >> 5;
    const int ln  = tid & 31;

    for (int idx = tid; idx < KK * (EE / 4); idx += 256) {
        const int r = idx / (EE / 4), c4 = (idx % (EE / 4)) << 2;
        *(float4*)(&rows[r * EE + c4]) =
            *(const float4*)(fc + (size_t)(c * KK + r) * EE + c4);
    }
    if (tid < KK) sinv[tid] = invnorm[c * KK + tid];
    __syncthreads();

    float wsum = 0.0f;
    for (int p = wv; p < 45; p += 8) {
        int i = 0, rem = p;
        while (rem >= 9 - i) { rem -= 9 - i; ++i; }
        const int j = i + 1 + rem;
        float d = 0.0f;
        for (int e = ln; e < EE; e += 32) d += rows[i * EE + e] * rows[j * EE + e];
#pragma unroll
        for (int m = 16; m >= 1; m >>= 1) d += __shfl_xor(d, m, 32);
        if (ln == 0) {
            d *= sinv[i] * sinv[j];
            float sub = 1.0f - d;
            sub = (sub <= 0.0f) ? 1e-10f : sub;
            wsum += sqrtf(2.0f * sub);
        }
    }
    if (ln == 0) atomicAdd(&acc[1], wsum);
}

// ---------------------------------------------------------------------------
__global__ __launch_bounds__(32) void st_combine_kernel(const float* __restrict__ acc,
                                                        float* __restrict__ out) {
    if (threadIdx.x == 0)
        out[0] = acc[0] + TAU * acc[1] / ((float)CC * (float)(KK * (KK - 1)));
}

// ---------------------------------------------------------------------------
extern "C" void kernel_launch(void* const* d_in, const int* in_sizes, int n_in,
                              void* d_out, int out_size, void* d_ws, size_t ws_size,
                              hipStream_t stream) {
    const float* emb    = (const float*)d_in[0];   // [256, 512]
    const int*   labels = (const int*)  d_in[1];   // [256]
    const float* fc     = (const float*)d_in[2];   // [81920, 512]
    float* ws     = (float*)d_ws;
    float* invn   = ws + WS_INV;
    float* logits = ws + WS_LOG;
    float* acc    = ws + WS_ACC;
    float* out    = (float*)d_out;

    st_zero_kernel<<<1, 32, 0, stream>>>(acc);
    st_invnorm_kernel<<<320, 256, 0, stream>>>(fc, invn);
    dim3 grid(CC / 8, BB / NBB);
    st_gemm_pool_kernel<<<grid, 320, 0, stream>>>(emb, labels, fc, invn, logits);
    st_ce_kernel<<<BB, 256, 0, stream>>>(logits, labels, acc);
    st_reg_kernel<<<CC, 256, 0, stream>>>(fc, invn, acc);
    st_combine_kernel<<<1, 32, 0, stream>>>(acc, out);
}
